// RelativeMultiHeadAttention_61907658604923
// MI455X (gfx1250) — matrix operand order repaired and checked by reading, not compiled, not run on hardware
//
#include <hip/hip_runtime.h>
#include <hip/hip_bf16.h>

// ---------------------------------------------------------------------------
// Transformer-XL relative multi-head attention for gfx1250 (MI455X).
// All matmuls in bf16 via v_wmma_f32_16x16x32_bf16 (f32 accumulate).
// Flash-attention style fused score/softmax/PV kernel (no score tensor).
// GEMM: 2-stage software-pipelined k-loop + global_prefetch.
// Attention: V chunks staged memory->LDS with global_load_async_to_lds_b128
// (ASYNCcnt), overlapping the score WMMAs and softmax.
// ---------------------------------------------------------------------------

typedef __bf16 bf16;
typedef __attribute__((ext_vector_type(16))) __bf16 v16bf;
typedef __attribute__((ext_vector_type(8)))  float  v8f;

static constexpr int DM = 512;   // d_model
static constexpr int NH = 8;     // heads
static constexpr int DK = 64;    // d_k
static constexpr int B_ = 4;     // batch
static constexpr int QL = 1024;  // q_len
static constexpr int ML = 1024;  // m_len
static constexpr int KL = 2048;  // k_len

// ----------------------------- helpers -------------------------------------

static __device__ inline unsigned short bf_bits(float f) {
  __bf16 h = (__bf16)f;
  return __builtin_bit_cast(unsigned short, h);
}
static __device__ inline __bf16 bits_bf(unsigned short u) {
  return __builtin_bit_cast(__bf16, u);
}

// A-fragment (16x32 bf16, M x K). Source row-major, row stride ld elements.
// Layout: lanes0-15 M=0..15, K=0..15 contiguous;
// lanes16-31: K=16..31.
static __device__ inline v16bf load_a_frag(const bf16* __restrict__ p, int ld, int lane) {
  const int m  = lane & 15;
  const int kb = (lane >> 4) << 4;  // 0 or 16
  const bf16* row = p + (size_t)m * ld;
  v16bf a;
#pragma unroll
  for (int i = 0; i < 4; ++i) {
    const int k = kb + 2 * i;
    a[2 * i]     = row[k];
    a[2 * i + 1] = row[k + 1];
  }
#pragma unroll
  for (int i = 4; i < 8; ++i) {
    const int k = kb + 2 * i;
    a[2 * i]     = row[k];
    a[2 * i + 1] = row[k + 1];
  }
  return a;
}

// B-fragment (32x16 bf16, K x N) where the source is stored as N rows of K
// contiguous elements (i.e. B transposed, row-major). lanes0-15: K=0..15,
// lanes16-31: K=16..31, column N = lane&15.
static __device__ inline v16bf load_b_frag(const bf16* __restrict__ p, int ld, int lane) {
  const int n  = lane & 15;
  const int kb = (lane >> 4) << 4;  // 0 or 16
  const bf16* row = p + (size_t)n * ld + kb;
  v16bf b;
#pragma unroll
  for (int e = 0; e < 16; ++e) b[e] = row[e];
  return b;
}

// Same as load_b_frag, but with explicit (clamped) source-row index per column.
static __device__ inline v16bf load_b_frag_clamp(const bf16* __restrict__ base, int ld,
                                                 int row0, int rmax, int coloff, int lane) {
  const int n  = lane & 15;
  const int kb = (lane >> 4) << 4;
  int rr = row0 + n;
  rr = rr < 0 ? 0 : (rr > rmax ? rmax : rr);
  const bf16* row = base + (size_t)rr * ld + coloff + kb;
  v16bf b;
#pragma unroll
  for (int e = 0; e < 16; ++e) b[e] = row[e];
  return b;
}

static __device__ inline v8f wmma_bf16(v16bf a, v16bf b, v8f c) {
  return __builtin_amdgcn_wmma_f32_16x16x32_bf16(false, a, false, b, (short)0, c, false, false);
}

// Async memory -> LDS, 16 bytes per lane. Writes LDS at lds_off,
// tracked by ASYNCcnt. lds_off is the low-32-bit LDS byte offset (flat-LDS
// aperture truncation of a generic __shared__ pointer).
static __device__ inline void async_load_lds_b128(unsigned lds_off, unsigned long long gaddr) {
  asm volatile("global_load_async_to_lds_b128 %0, %1, off"
               :: "v"(lds_off), "v"(gaddr)
               : "memory");
}
static __device__ inline void wait_asynccnt0() {
  asm volatile("s_wait_asynccnt 0x0" ::: "memory");
}

// ----------------------------- pack kernels --------------------------------

// x = concat(inputs_mem, inputs) along seq, f32 -> bf16. Total B_*KL*DM elems.
__global__ void pack_x_kernel(const float* __restrict__ inp, const float* __restrict__ mem,
                              bf16* __restrict__ xb) {
  size_t i = (size_t)blockIdx.x * blockDim.x + threadIdx.x;
  const int d   = (int)(i & (DM - 1));
  const size_t row = i >> 9;               // b*KL + t
  const int t = (int)(row & (KL - 1));
  const int b = (int)(row >> 11);
  float val = (t < ML) ? mem[((size_t)b * ML + t) * DM + d]
                       : inp[((size_t)b * QL + (t - ML)) * DM + d];
  xb[i] = (bf16)val;
}

// dst[n][k] = (bf16) src[k][n]   (src: K x N row-major f32)
__global__ void transpose_to_bf16(const float* __restrict__ src, bf16* __restrict__ dst,
                                  int K, int N) {
  size_t i = (size_t)blockIdx.x * blockDim.x + threadIdx.x;
  if (i >= (size_t)K * N) return;
  const int n = (int)(i % N);
  const int k = (int)(i / N);
  dst[(size_t)n * K + k] = (bf16)src[i];
}

__global__ void convert_to_bf16(const float* __restrict__ src, bf16* __restrict__ dst, size_t n) {
  size_t i = (size_t)blockIdx.x * blockDim.x + threadIdx.x;
  if (i < n) dst[i] = (bf16)src[i];
}

// qu = heads_q + u ; qv = heads_q + v   (bf16). Total B_*QL*DM elems.
__global__ void build_q_kernel(const bf16* __restrict__ wqkv, const float* __restrict__ u,
                               const float* __restrict__ v, bf16* __restrict__ qu,
                               bf16* __restrict__ qv) {
  size_t i = (size_t)blockIdx.x * blockDim.x + threadIdx.x;
  const int d = (int)(i & (DM - 1));       // h*64 + dk
  const size_t row = i >> 9;               // b*QL + q
  const int q = (int)(row & (QL - 1));
  const int b = (int)(row >> 10);
  const float w = (float)wqkv[((size_t)b * KL + ML + q) * (3 * DM) + d];
  qu[i] = (bf16)(w + u[d]);
  qv[i] = (bf16)(w + v[d]);
}

// ----------------------------- GEMM kernel ---------------------------------
// C[M,N] = A[M,K] @ Bt[N,K]^T + bias[N].  One wave -> 16x64 output tile.
// Exactly (M/16)*(N/64) waves, 4 waves/block. K must be a multiple of 64.
// Two-stage software pipeline: loads for step s+1 issue before the WMMAs of
// step s, so the matrix pipe overlaps memory latency instead of stalling on
// s_wait_loadcnt before every wmma.
__global__ __launch_bounds__(128) void gemm_bf16_wmma(
    const bf16* __restrict__ A, const bf16* __restrict__ Bt,
    const float* __restrict__ bias, bf16* __restrict__ Cb, float* __restrict__ Cf,
    int M, int N, int K) {
  const int wave = (int)((blockIdx.x * blockDim.x + threadIdx.x) >> 5);
  const int lane = (int)(threadIdx.x & 31);
  const int nt64 = N >> 6;
  const int mt = wave / nt64;
  const int nt = wave % nt64;

  const bf16* Arow  = A  + (size_t)mt * 16 * K;
  const bf16* Bbase = Bt + (size_t)nt * 64 * K;

  v8f acc[4] = {};

  // Stage 0 preload (kk = 0).
  v16bf a0 = load_a_frag(Arow, K, lane);
  v16bf B0[4];
#pragma unroll
  for (int t = 0; t < 4; ++t) B0[t] = load_b_frag(Bbase + (size_t)t * 16 * K, K, lane);

  for (int kk = 0; kk < K; kk += 64) {
    // Prefetch two steps ahead into L2/WGP$ (global_prefetch_b8).
    if (kk + 128 < K) {
      __builtin_prefetch(Arow + kk + 128, 0, 1);
#pragma unroll
      for (int t = 0; t < 4; ++t)
        __builtin_prefetch(Bbase + (size_t)t * 16 * K + kk + 128, 0, 1);
    }
    // Issue stage-1 loads (kk+32), then do stage-0 math.
    v16bf a1 = load_a_frag(Arow + kk + 32, K, lane);
    v16bf B1[4];
#pragma unroll
    for (int t = 0; t < 4; ++t)
      B1[t] = load_b_frag(Bbase + (size_t)t * 16 * K + kk + 32, K, lane);
#pragma unroll
    for (int t = 0; t < 4; ++t) acc[t] = wmma_bf16(a0, B0[t], acc[t]);

    // Issue next stage-0 loads (kk+64), then do stage-1 math.
    if (kk + 64 < K) {
      a0 = load_a_frag(Arow + kk + 64, K, lane);
#pragma unroll
      for (int t = 0; t < 4; ++t)
        B0[t] = load_b_frag(Bbase + (size_t)t * 16 * K + kk + 64, K, lane);
    }
#pragma unroll
    for (int t = 0; t < 4; ++t) acc[t] = wmma_bf16(a1, B1[t], acc[t]);
  }

  const int hl = lane >> 4, nl = lane & 15;
#pragma unroll
  for (int t = 0; t < 4; ++t) {
#pragma unroll
    for (int r = 0; r < 8; ++r) {
      const int m = r + 8 * hl;
      const int col = nt * 64 + t * 16 + nl;
      const float cv = acc[t][r] + bias[col];
      const size_t off = (size_t)(mt * 16 + m) * N + col;
      if (Cb) Cb[off] = (bf16)cv;
      else    Cf[off] = cv;
    }
  }
}

// --------------------------- attention kernel ------------------------------
// One wave per (b, h, 16-row q tile). Online softmax over 32-key chunks.
// bd relative shift: bd[i, j] = bd'[i, j + QL-1 - i]; per 16-key subtile the
// needed bd' columns form a 31-wide band -> compute 2 unshifted WMMA tiles,
// stage to LDS, gather the skewed diagonal.
__global__ __launch_bounds__(128) void attn_wmma(
    const bf16* __restrict__ qu, const bf16* __restrict__ qv,
    const bf16* __restrict__ wqkv, const bf16* __restrict__ rproj,
    bf16* __restrict__ outb) {
  __shared__ float          s_bd[4][16 * 32];
  __shared__ unsigned short s_p[4][16 * 32];
  __shared__ alignas(16) unsigned short s_v[4][32 * 64];

  const int wib  = (int)(threadIdx.x >> 5);
  const int lane = (int)(threadIdx.x & 31);
  const int wave = (int)blockIdx.x * 4 + wib;   // 0..2047
  const int qt = wave & 63;
  const int h  = (wave >> 6) & 7;
  const int b  = wave >> 9;
  const int q0 = qt << 4;
  const int hl = lane >> 4, nl = lane & 15;

  const bf16* Qu = qu + (size_t)b * QL * DM + h * DK;             // ld = DM
  const bf16* Qv = qv + (size_t)b * QL * DM + h * DK;             // ld = DM
  const bf16* Kp = wqkv + (size_t)b * KL * (3 * DM) + DM + h * DK;     // ld = 3*DM
  const bf16* Vp = wqkv + (size_t)b * KL * (3 * DM) + 2 * DM + h * DK; // ld = 3*DM
  const bf16* Rp = rproj + h * DK;                                // ld = DM

  // LDS byte offset of this wave's V staging buffer (low 32 bits of the
  // generic address = LDS offset in the flat-LDS aperture).
  const unsigned vbase = (unsigned)(unsigned long long)(const void*)&s_v[wib][0];

  // Q fragments (16 rows x 64 d), resident for the whole loop.
  const v16bf aU0 = load_a_frag(Qu + (size_t)q0 * DM,      DM, lane);
  const v16bf aU1 = load_a_frag(Qu + (size_t)q0 * DM + 32, DM, lane);
  const v16bf aV0 = load_a_frag(Qv + (size_t)q0 * DM,      DM, lane);
  const v16bf aV1 = load_a_frag(Qv + (size_t)q0 * DM + 32, DM, lane);

  v8f o[4] = {};
  float mrow[8], lrow[8];
#pragma unroll
  for (int r = 0; r < 8; ++r) { mrow[r] = -3.0e38f; lrow[r] = 0.0f; }

  const int nch = (q0 + QL + 16 + 31) >> 5;  // ceil((q0 + 1040)/32); nch*32 <= KL

  for (int ch = 0; ch < nch; ++ch) {
    const int k0c = ch << 5;

    // Async-stage V chunk (32 keys x 64 d = 4 KB) memory -> LDS. Each lane
    // owns one key row (128 B = 8 x b128). Overlaps the AC/BD WMMAs and the
    // softmax below; waited on just before the PV fragment reads.
    {
      const unsigned long long gv =
          (unsigned long long)(const void*)(Vp + (size_t)(k0c + lane) * (3 * DM));
      const unsigned lv = vbase + (unsigned)(lane * 128);
#pragma unroll
      for (int j = 0; j < 8; ++j)
        async_load_lds_b128(lv + j * 16, gv + j * 16);
    }
    // Prefetch next chunk's K rows (one byte per row pulls the cacheline).
    if (ch + 1 < nch)
      __builtin_prefetch(Kp + (size_t)(k0c + 32 + lane) * (3 * DM), 0, 1);

    float pv[2][8];
    float cmax[8];
#pragma unroll
    for (int r = 0; r < 8; ++r) cmax[r] = -3.0e38f;

#pragma unroll
    for (int s = 0; s < 2; ++s) {
      const int k0 = k0c + s * 16;

      // ---- AC tile: (q+u)[16x64] @ K^T[64x16] -> 2 WMMAs
      v8f sAcc = {};
      {
        v16bf bk0 = load_b_frag(Kp + (size_t)k0 * (3 * DM),      3 * DM, lane);
        v16bf bk1 = load_b_frag(Kp + (size_t)k0 * (3 * DM) + 32, 3 * DM, lane);
        sAcc = wmma_bf16(aU0, bk0, sAcc);
        sAcc = wmma_bf16(aU1, bk1, sAcc);
      }

      // ---- BD band: bd'[i, c0 + 0..31], c0 = k0 + (QL-16) - q0
      const int c0 = k0 + (QL - 16) - q0;
#pragma unroll
      for (int tt = 0; tt < 2; ++tt) {
        v8f bdAcc = {};
        v16bf br0 = load_b_frag_clamp(Rp, DM, c0 + tt * 16, KL - 1, 0,  lane);
        v16bf br1 = load_b_frag_clamp(Rp, DM, c0 + tt * 16, KL - 1, 32, lane);
        bdAcc = wmma_bf16(aV0, br0, bdAcc);
        bdAcc = wmma_bf16(aV1, br1, bdAcc);
#pragma unroll
        for (int r = 0; r < 8; ++r) {
          const int m = r + 8 * hl;
          s_bd[wib][m * 32 + tt * 16 + nl] = bdAcc[r];
        }
      }

      // ---- combine (skewed gather), scale, causal mask
#pragma unroll
      for (int r = 0; r < 8; ++r) {
        const int m  = r + 8 * hl;
        const int oo = nl - m + 15;              // 0..30
        float sc = (sAcc[r] + s_bd[wib][m * 32 + oo]) * 0.125f;  // 1/sqrt(64)
        const int kj = k0 + nl;
        const int qi = q0 + m;
        if (kj > qi + ML) sc = -1.0e30f;
        pv[s][r] = sc;
        cmax[r] = fmaxf(cmax[r], sc);
      }
    }

    // ---- online softmax update (row stats live in each 16-lane half)
#pragma unroll
    for (int r = 0; r < 8; ++r) {
      float mx = cmax[r];
#pragma unroll
      for (int msk = 1; msk < 16; msk <<= 1) mx = fmaxf(mx, __shfl_xor(mx, msk, 32));
      const float nm = fmaxf(mrow[r], mx);
      const float al = __expf(mrow[r] - nm);
      mrow[r] = nm;
      const float p0 = __expf(pv[0][r] - nm);
      const float p1 = __expf(pv[1][r] - nm);
      pv[0][r] = p0;
      pv[1][r] = p1;
      float sm = p0 + p1;
#pragma unroll
      for (int msk = 1; msk < 16; msk <<= 1) sm += __shfl_xor(sm, msk, 32);
      lrow[r] = lrow[r] * al + sm;
      o[0][r] *= al; o[1][r] *= al; o[2][r] *= al; o[3][r] *= al;
    }

    // ---- stage P (16x32) as bf16, rebuild A-fragment layout from LDS
#pragma unroll
    for (int s = 0; s < 2; ++s) {
#pragma unroll
      for (int r = 0; r < 8; ++r) {
        const int m = r + 8 * hl;
        s_p[wib][m * 32 + s * 16 + nl] = bf_bits(pv[s][r]);
      }
    }
    v16bf ap;
    {
      const int m  = lane & 15;
      const int kb = hl << 4;
#pragma unroll
      for (int i = 0; i < 8; ++i) {
        const int k = kb + 2 * i;
        ap[2 * i]     = bits_bf(s_p[wib][m * 32 + k]);
        ap[2 * i + 1] = bits_bf(s_p[wib][m * 32 + k + 1]);
      }
    }

    // ---- PV: P[16x32] @ V[32x64] -> 4 WMMAs (B read transposed from LDS)
    wait_asynccnt0();   // V chunk resident in LDS
    const int kb16 = hl << 4;
#pragma unroll
    for (int t = 0; t < 4; ++t) {
      v16bf bv;
#pragma unroll
      for (int e = 0; e < 16; ++e)
        bv[e] = bits_bf(s_v[wib][(kb16 + e) * 64 + t * 16 + nl]);
      o[t] = wmma_bf16(ap, bv, o[t]);
    }
  }

  // ---- epilogue: normalize, store bf16 (4096 x 512 row-major)
#pragma unroll
  for (int r = 0; r < 8; ++r) {
    const int m = r + 8 * hl;
    const float inv = 1.0f / lrow[r];
    const size_t row = (size_t)(b * QL + q0 + m);
#pragma unroll
    for (int t = 0; t < 4; ++t)
      outb[row * DM + h * DK + t * 16 + nl] = (bf16)(o[t][r] * inv);
  }
}

// ----------------------------- launch --------------------------------------

extern "C" void kernel_launch(void* const* d_in, const int* in_sizes, int n_in,
                              void* d_out, int out_size, void* d_ws, size_t ws_size,
                              hipStream_t stream) {
  const float* inputs     = (const float*)d_in[0];
  const float* inputs_mem = (const float*)d_in[1];
  const float* r          = (const float*)d_in[2];
  const float* W_qkv      = (const float*)d_in[3];
  const float* b_qkv      = (const float*)d_in[4];
  const float* W_r        = (const float*)d_in[5];
  const float* b_r        = (const float*)d_in[6];
  const float* W_o        = (const float*)d_in[7];
  const float* b_o        = (const float*)d_in[8];
  const float* u          = (const float*)d_in[9];
  const float* v          = (const float*)d_in[10];
  float* out = (float*)d_out;

  // Workspace layout (bf16), total ~50.5 MB.
  char* w = (char*)d_ws;
  auto take = [&](size_t elems) { bf16* p = (bf16*)w; w += elems * sizeof(bf16); return p; };
  bf16* xb     = take((size_t)B_ * KL * DM);       // concat input, bf16
  bf16* wqkvT  = take((size_t)3 * DM * DM);        // W_qkv^T
  bf16* wrT    = take((size_t)DM * DM);            // W_r^T
  bf16* woT    = take((size_t)DM * DM);            // W_o^T
  bf16* rb     = take((size_t)KL * DM);            // r, bf16
  bf16* qkvb   = take((size_t)B_ * KL * 3 * DM);   // QKV projections
  bf16* rprojb = take((size_t)KL * DM);            // r projection
  bf16* qub    = take((size_t)B_ * QL * DM);       // q + u
  bf16* qvb    = take((size_t)B_ * QL * DM);       // q + v
  bf16* ob     = take((size_t)B_ * QL * DM);       // attention output

  const dim3 blk(256);
  pack_x_kernel<<<(B_ * KL * DM) / 256, blk, 0, stream>>>(inputs, inputs_mem, xb);
  transpose_to_bf16<<<(3 * DM * DM) / 256, blk, 0, stream>>>(W_qkv, wqkvT, DM, 3 * DM);
  transpose_to_bf16<<<(DM * DM) / 256, blk, 0, stream>>>(W_r, wrT, DM, DM);
  transpose_to_bf16<<<(DM * DM) / 256, blk, 0, stream>>>(W_o, woT, DM, DM);
  convert_to_bf16<<<(KL * DM) / 256, blk, 0, stream>>>(r, rb, (size_t)KL * DM);

  // QKV: M=8192 N=1536 K=512 -> (512*24)=12288 waves
  gemm_bf16_wmma<<<12288 / 4, 128, 0, stream>>>(xb, wqkvT, b_qkv, qkvb, nullptr,
                                                B_ * KL, 3 * DM, DM);
  // R proj: M=2048 N=512 K=512 -> 1024 waves
  gemm_bf16_wmma<<<1024 / 4, 128, 0, stream>>>(rb, wrT, b_r, rprojb, nullptr, KL, DM, DM);

  build_q_kernel<<<(B_ * QL * DM) / 256, blk, 0, stream>>>(qkvb, u, v, qub, qvb);

  // Fused attention: 4*8*64 = 2048 waves
  attn_wmma<<<2048 / 4, 128, 0, stream>>>(qub, qvb, qkvb, rprojb, ob);

  // Out proj (f32 result): M=4096 N=512 K=512 -> 2048 waves
  gemm_bf16_wmma<<<2048 / 4, 128, 0, stream>>>(ob, woT, b_o, nullptr, out, B_ * QL, DM, DM);

  (void)in_sizes; (void)n_in; (void)out_size; (void)ws_size;
}
